// CAB_85950885527964
// MI455X (gfx1250) — matrix operand order repaired and checked by reading, not compiled, not run on hardware
//
#include <hip/hip_runtime.h>

typedef __attribute__((ext_vector_type(8)))  _Float16 v8h;
typedef __attribute__((ext_vector_type(16))) _Float16 v16h;
typedef __attribute__((ext_vector_type(8)))  float    v8f;

#define EPSV 1e-5f
#define SCALE_ATT 0.17677669529663687f

__device__ __forceinline__ float mishf(float x){
  float sp = (x > 20.f) ? x : log1pf(__expf(x));
  return x * tanhf(sp);
}

__device__ __forceinline__ v16h cat16(v8h lo, v8h hi){
  return __builtin_shufflevector(lo, hi, 0,1,2,3,4,5,6,7,8,9,10,11,12,13,14,15);
}

// ---------------------------------------------------------------------------
// Per-channel BN folding: chan[0:256)=scaleAn, [256)=shiftAn (aconv_b folded),
// [512)=inv1, [768)=sh1, [1024)=inv2, [1280)=sh2
// ---------------------------------------------------------------------------
__global__ void k_prep_chan(const float* an_g, const float* an_b, const float* an_m, const float* an_v,
                            const float* aconv_b,
                            const float* n1_g, const float* n1_b, const float* n1_m, const float* n1_v,
                            const float* n2_g, const float* n2_b, const float* n2_m, const float* n2_v,
                            float* chan){
  int c = threadIdx.x;
  float ia = an_g[c] * rsqrtf(an_v[c] + EPSV);
  chan[c]       = ia;
  chan[256 + c] = an_b[c] + (aconv_b[c] - an_m[c]) * ia;
  float i1 = n1_g[c] * rsqrtf(n1_v[c] + EPSV);
  chan[512 + c] = i1;
  chan[768 + c] = n1_b[c] - n1_m[c] * i1;
  float i2 = n2_g[c] * rsqrtf(n2_v[c] + EPSV);
  chan[1024 + c] = i2;
  chan[1280 + c] = n2_b[c] - n2_m[c] * i2;
}

// ---------------------------------------------------------------------------
// Weight conversion + WMMA B-fragment swizzle (contiguous k order: k = 16*half + i, same as the A loads).
// d[((nt*kT + kt)*32 + lane)*16 + i]: col = nt*16 + (lane&15),
// K = kt*32 + (lane>=16 ? 16 : 0) + i.
// mode 0: src [K,N]; mode 1: src [N,K]; mode 2: aconv OIHW (256,256,2,2)
// ---------------------------------------------------------------------------
__global__ void k_swz(const float* __restrict__ s, _Float16* __restrict__ d,
                      int K, int N, int mode){
  int e = blockIdx.x * 256 + threadIdx.x;
  if (e >= K * N) return;
  int i    = e & 15;
  int lane = (e >> 4) & 31;
  int tile = e >> 9;
  int kT = K >> 5;
  int nt = tile / kT, kt = tile - nt * kT;
  int col = nt * 16 + (lane & 15);
  int kk  = kt * 32 + ((lane >> 4) << 4) + i;
  float v;
  if (mode == 0)      v = s[(size_t)kk * N + col];
  else if (mode == 1) v = s[(size_t)col * K + kk];
  else                v = s[(size_t)col * 1024 + (kk & 255) * 4 + (kk >> 8)];
  d[e] = (_Float16)v;
}

// ------------------------- BN1 -> NHWC f16 ---------------------------------
__global__ void k_bn1(const float* __restrict__ x, const float* __restrict__ chan,
                      _Float16* __restrict__ xh, int n){
  int e = blockIdx.x * 256 + threadIdx.x;
  if (e >= n) return;
  int b = e >> 20, c = (e >> 12) & 255, hw = e & 4095;
  float v = x[e] * chan[512 + c] + chan[768 + c];
  xh[((size_t)((b << 12) + hw)) * 256 + c] = (_Float16)v;
}

// ------------------------- im2col for 2x2 stride-2 conv --------------------
__global__ void k_im2col(const _Float16* __restrict__ xh, _Float16* __restrict__ axc, int n){
  int e = blockIdx.x * 256 + threadIdx.x;
  if (e >= n) return;
  int m = e >> 10, j = e & 1023;
  int p = j >> 8, cc = j & 255;
  int b = m >> 10, hd = (m >> 5) & 31, wd = m & 31;
  int t = (b * 64 + 2 * hd + (p >> 1)) * 64 + 2 * wd + (p & 1);
  axc[e] = xh[(size_t)t * 256 + cc];
}

// ---------------------------------------------------------------------------
// Generic WMMA GEMM: C[M,N] = A[M,K] * B[K,N], f16 in, f32 accumulate.
// A row-major; B pre-swizzled. Wave tile 16M x 64N, block 64M x 128N.
// ---------------------------------------------------------------------------
struct GemmP {
  const _Float16* A; int lda;
  const _Float16* B;               // swizzled
  int M, N, K;
  int mode;        // bit0: gather A rows from central region; bit1: scatter out rows
  int epi;         // 0: f16; 1: f16 scale/shift; 2: f16 +bias; 3: f32 +bias +resid -> NCHW
  _Float16* out16; int ldo;
  float* out32;
  const float* scaleN; const float* shiftN; const float* biasN;
  const float* resid;
};

__device__ __forceinline__ int central_row(int m){
  int b = m >> 10, hc = (m >> 5) & 31, wc = m & 31;
  return b * 4096 + (hc + 16) * 64 + (wc + 16);
}

__global__ void k_gemm(GemmP p){
  int wave = threadIdx.x >> 5;
  int lane = threadIdx.x & 31;
  int half_ = lane >> 4;
  int r = lane & 15;
  int m_off = blockIdx.x * 64 + (wave & 3) * 16;
  int n_off = blockIdx.y * 128 + (wave >> 2) * 64;
  int nt0 = n_off >> 4;
  int kT = p.K >> 5;

  int arow = m_off + r;
  if (p.mode & 1) arow = central_row(arow);
  const _Float16* Arow = p.A + (size_t)arow * p.lda;

  v8f acc[4] = {};
  for (int k = 0; k < p.K; k += 32){
    int kt = k >> 5;
    v8h alo = *(const v8h*)(Arow + k + (half_ ? 16 : 0));
    v8h ahi = *(const v8h*)(Arow + k + (half_ ? 24 : 8));
    v16h af = cat16(alo, ahi);
    __builtin_prefetch(Arow + k + 32, 0, 3);      // speculative; OOB dropped
#pragma unroll
    for (int t = 0; t < 4; t++){
      const _Float16* bp = p.B + (((size_t)(nt0 + t) * kT + kt) * 32 + lane) * 16;
      v8h blo = *(const v8h*)bp;
      v8h bhi = *(const v8h*)(bp + 8);
      v16h bf = cat16(blo, bhi);
      __builtin_prefetch(bp + 512, 0, 3);
      acc[t] = __builtin_amdgcn_wmma_f32_16x16x32_f16(false, af, false, bf,
                                                      (short)0, acc[t], false, false);
    }
  }

#pragma unroll
  for (int t = 0; t < 4; t++){
    int col = n_off + t * 16 + r;
#pragma unroll
    for (int rv = 0; rv < 8; rv++){
      int row = m_off + rv + (half_ ? 8 : 0);
      float v = acc[t][rv];
      int orow = (p.mode & 2) ? central_row(row) : row;
      if (p.epi == 0){
        p.out16[(size_t)orow * p.ldo + col] = (_Float16)v;
      } else if (p.epi == 1){
        p.out16[(size_t)orow * p.ldo + col] = (_Float16)(v * p.scaleN[col] + p.shiftN[col]);
      } else if (p.epi == 2){
        p.out16[(size_t)orow * p.ldo + col] = (_Float16)(v + p.biasN[col]);
      } else {
        float s = v + p.biasN[col] + p.resid[(size_t)row * p.ldo + col];
        int b = row >> 12, hw = row & 4095;
        p.out32[((size_t)(b * 256 + col)) * 4096 + hw] = s;   // NCHW
      }
    }
  }
}

// ---------------------------------------------------------------------------
// WMMA windowed MHA, global path: per (window, head): 64 q x 16 kv (pad to 32).
// Block = 1 window, 8 waves = 8 heads. Wave-private LDS (no barriers needed:
// same-wave DS ops are in-order). svt = V^T zero-padded [32][40], sp = P [64][40].
// ---------------------------------------------------------------------------
__global__ void __launch_bounds__(256) k_mha_g(const _Float16* __restrict__ q,
                                               const _Float16* __restrict__ kv,
                                               _Float16* __restrict__ out){
  __shared__ _Float16 smem[8][3840];        // 60 KB
  int wave = threadIdx.x >> 5, lane = threadIdx.x & 31;
  int head = wave;
  int win = blockIdx.x;
  int b = win >> 6, wh = (win >> 3) & 7, ww = win & 7;
  _Float16* svt = smem[wave];               // [32][40]
  _Float16* sp  = smem[wave] + 1280;        // [64][40]
  int half_ = lane >> 4, r = lane & 15;
  int dbase = (b * 32 + wh * 4) * 32 + ww * 4;   // dtok(j) = dbase + (j>>2)*32 + (j&3)
  int qbase = (b * 64 + wh * 8) * 64 + ww * 8;   // qtok(i) = qbase + (i>>3)*64 + (i&7)

  for (int idx = lane; idx < 1280; idx += 32) svt[idx] = (_Float16)0.f;
  for (int idx = lane; idx < 512; idx += 32){    // V^T: d = lane, j = idx>>5
    int j = idx >> 5;
    int dt = dbase + (j >> 2) * 32 + (j & 3);
    svt[lane * 40 + j] = kv[(size_t)dt * 512 + 256 + head * 32 + lane];
  }

  // S = Q K^T (one 16-col tile), softmax in registers, P -> sp (cols 16..31 = 0)
  int dtb = dbase + (r >> 2) * 32 + (r & 3);     // this lane's key token (col r)
  const _Float16* kp = kv + (size_t)dtb * 512 + head * 32 + (half_ ? 16 : 0);
  v8h klo = *(const v8h*)kp;
  v8h khi = *(const v8h*)(kp + 8);
  v16h bfk = cat16(klo, khi);
#pragma unroll
  for (int mt = 0; mt < 4; mt++){
    int qrow = mt * 16 + r;
    int qt = qbase + (qrow >> 3) * 64 + (qrow & 7);
    const _Float16* qp = q + (size_t)qt * 256 + head * 32;
    v8h alo = *(const v8h*)(qp + (half_ ? 16 : 0));
    v8h ahi = *(const v8h*)(qp + (half_ ? 24 : 8));
    v16h af = cat16(alo, ahi);
    v8f z = {};
    v8f s = __builtin_amdgcn_wmma_f32_16x16x32_f16(false, af, false, bfk,
                                                   (short)0, z, false, false);
#pragma unroll
    for (int rv = 0; rv < 8; rv++){
      float v = s[rv] * SCALE_ATT;
      float mx = v;
#pragma unroll
      for (int o1 = 1; o1 < 16; o1 <<= 1) mx = fmaxf(mx, __shfl_xor(mx, o1, 32));
      float pv = __expf(v - mx);
      float sum = pv;
#pragma unroll
      for (int o1 = 1; o1 < 16; o1 <<= 1) sum += __shfl_xor(sum, o1, 32);
      int row = mt * 16 + rv + (half_ ? 8 : 0);
      sp[row * 40 + r]      = (_Float16)(pv / sum);
      sp[row * 40 + 16 + r] = (_Float16)0.f;
    }
  }

  // O = P V  (K padded to 32)
  v8f oacc[4][2] = {};
#pragma unroll
  for (int mt = 0; mt < 4; mt++){
    const _Float16* ap = sp + (mt * 16 + r) * 40;
    v8h alo = *(const v8h*)(ap + (half_ ? 16 : 0));
    v8h ahi = *(const v8h*)(ap + (half_ ? 24 : 8));
    v16h af = cat16(alo, ahi);
#pragma unroll
    for (int nt = 0; nt < 2; nt++){
      const _Float16* bp = svt + (nt * 16 + r) * 40 + (half_ ? 16 : 0);
      v8h blo = *(const v8h*)bp;
      v8h bhi = *(const v8h*)(bp + 8);
      v16h bf = cat16(blo, bhi);
      oacc[mt][nt] = __builtin_amdgcn_wmma_f32_16x16x32_f16(false, af, false, bf,
                                                            (short)0, oacc[mt][nt], false, false);
    }
  }
#pragma unroll
  for (int mt = 0; mt < 4; mt++)
#pragma unroll
    for (int nt = 0; nt < 2; nt++)
#pragma unroll
      for (int rv = 0; rv < 8; rv++){
        int row = mt * 16 + rv + (half_ ? 8 : 0);
        int qt = qbase + (row >> 3) * 64 + (row & 7);
        out[(size_t)qt * 256 + head * 32 + nt * 16 + r] = (_Float16)oacc[mt][nt][rv];
      }
}

// ---------------------------------------------------------------------------
// WMMA windowed MHA, central path: per (window, head): 64 q x 64 kv.
// Block = 128 threads = 4 heads; grid = windows * 2.
// svt = V^T [32][72]; sp = P [64][72].
// ---------------------------------------------------------------------------
__global__ void __launch_bounds__(128) k_mha_c(const _Float16* __restrict__ qkv,
                                               _Float16* __restrict__ out){
  __shared__ _Float16 smem[4][6912];        // 54 KB
  int wave = threadIdx.x >> 5, lane = threadIdx.x & 31;
  int head = (blockIdx.x & 1) * 4 + wave;
  int win  = blockIdx.x >> 1;
  int b = win >> 4, wh = (win >> 2) & 3, ww = win & 3;
  _Float16* svt = smem[wave];               // [32][72]
  _Float16* sp  = smem[wave] + 2304;        // [64][72]
  int half_ = lane >> 4, r = lane & 15;
  int base = (b * 32 + wh * 8) * 32 + ww * 8;    // ctok(j) = base + (j>>3)*32 + (j&7)

  for (int idx = lane; idx < 2048; idx += 32){   // V^T: d = lane, j = idx>>5
    int j = idx >> 5;
    int ct = base + (j >> 3) * 32 + (j & 7);
    svt[lane * 72 + j] = qkv[(size_t)ct * 768 + 512 + head * 32 + lane];
  }

  // S = Q K^T row-block by row-block; softmax in registers; P -> sp
#pragma unroll
  for (int mt = 0; mt < 4; mt++){
    int qrow = mt * 16 + r;
    int ctq = base + (qrow >> 3) * 32 + (qrow & 7);
    const _Float16* qp = qkv + (size_t)ctq * 768 + head * 32;
    v8h alo = *(const v8h*)(qp + (half_ ? 16 : 0));
    v8h ahi = *(const v8h*)(qp + (half_ ? 24 : 8));
    v16h af = cat16(alo, ahi);
    v8f s[4];
#pragma unroll
    for (int nt = 0; nt < 4; nt++){
      int kj = nt * 16 + r;
      int ctk = base + (kj >> 3) * 32 + (kj & 7);
      const _Float16* kp = qkv + (size_t)ctk * 768 + 256 + head * 32 + (half_ ? 16 : 0);
      v8h blo = *(const v8h*)kp;
      v8h bhi = *(const v8h*)(kp + 8);
      v16h bf = cat16(blo, bhi);
      v8f z = {};
      s[nt] = __builtin_amdgcn_wmma_f32_16x16x32_f16(false, af, false, bf,
                                                     (short)0, z, false, false);
    }
#pragma unroll
    for (int rv = 0; rv < 8; rv++){
      float v0 = s[0][rv] * SCALE_ATT, v1 = s[1][rv] * SCALE_ATT;
      float v2 = s[2][rv] * SCALE_ATT, v3 = s[3][rv] * SCALE_ATT;
      float mx = fmaxf(fmaxf(v0, v1), fmaxf(v2, v3));
#pragma unroll
      for (int o1 = 1; o1 < 16; o1 <<= 1) mx = fmaxf(mx, __shfl_xor(mx, o1, 32));
      v0 = __expf(v0 - mx); v1 = __expf(v1 - mx);
      v2 = __expf(v2 - mx); v3 = __expf(v3 - mx);
      float sum = v0 + v1 + v2 + v3;
#pragma unroll
      for (int o1 = 1; o1 < 16; o1 <<= 1) sum += __shfl_xor(sum, o1, 32);
      float inv = 1.f / sum;
      int row = mt * 16 + rv + (half_ ? 8 : 0);
      sp[row * 72 + r]      = (_Float16)(v0 * inv);
      sp[row * 72 + 16 + r] = (_Float16)(v1 * inv);
      sp[row * 72 + 32 + r] = (_Float16)(v2 * inv);
      sp[row * 72 + 48 + r] = (_Float16)(v3 * inv);
    }
  }

  // O = P V  (K = 64, two k-steps)
  v8f oacc[4][2] = {};
#pragma unroll
  for (int mt = 0; mt < 4; mt++){
#pragma unroll
    for (int kt = 0; kt < 2; kt++){
      const _Float16* ap = sp + (mt * 16 + r) * 72 + kt * 32;
      v8h alo = *(const v8h*)(ap + (half_ ? 16 : 0));
      v8h ahi = *(const v8h*)(ap + (half_ ? 24 : 8));
      v16h af = cat16(alo, ahi);
#pragma unroll
      for (int nt = 0; nt < 2; nt++){
        const _Float16* bp = svt + (nt * 16 + r) * 72 + kt * 32 + (half_ ? 16 : 0);
        v8h blo = *(const v8h*)bp;
        v8h bhi = *(const v8h*)(bp + 8);
        v16h bf = cat16(blo, bhi);
        oacc[mt][nt] = __builtin_amdgcn_wmma_f32_16x16x32_f16(false, af, false, bf,
                                                              (short)0, oacc[mt][nt], false, false);
      }
    }
  }
#pragma unroll
  for (int mt = 0; mt < 4; mt++)
#pragma unroll
    for (int nt = 0; nt < 2; nt++)
#pragma unroll
      for (int rv = 0; rv < 8; rv++){
        int row = mt * 16 + rv + (half_ ? 8 : 0);
        int ct = base + (row >> 3) * 32 + (row & 7);
        out[(size_t)ct * 256 + head * 32 + nt * 16 + r] = (_Float16)oacc[mt][nt][rv];
      }
}

// ------------------------- residual + BN2 ----------------------------------
__global__ void k_resid_bn2(const float* __restrict__ x, const _Float16* __restrict__ attn,
                            const float* __restrict__ chan,
                            float* __restrict__ x2, _Float16* __restrict__ xn2, int n){
  int e = blockIdx.x * 256 + threadIdx.x;
  if (e >= n) return;
  int m = e >> 8, c = e & 255;
  int b = m >> 12, hw = m & 4095;
  float s = x[((size_t)(b * 256 + c) << 12) + hw] + (float)attn[e];
  x2[e] = s;
  xn2[e] = (_Float16)(s * chan[1024 + c] + chan[1280 + c]);
}

// ------------------------- depthwise 3x3 + mish ----------------------------
__global__ void k_dw(const _Float16* __restrict__ h1, const float* __restrict__ dw_w,
                     const float* __restrict__ dw_b, _Float16* __restrict__ h2, int n){
  int e = blockIdx.x * 256 + threadIdx.x;
  if (e >= n) return;
  int m = e >> 10, f = e & 1023;
  int b = m >> 12, hw = m & 4095;
  int h = hw >> 6, w = hw & 63;
  float acc = dw_b[f];
#pragma unroll
  for (int dh = 0; dh < 3; dh++){
    int hh = h + dh - 1;
    if (hh < 0 || hh > 63) continue;
#pragma unroll
    for (int dw = 0; dw < 3; dw++){
      int wn = w + dw - 1;
      if (wn < 0 || wn > 63) continue;
      acc += (float)h1[((size_t)((b << 12) + hh * 64 + wn)) * 1024 + f] * dw_w[f * 9 + dh * 3 + dw];
    }
  }
  h2[e] = (_Float16)mishf(acc);
}

// ------------------------- dec scalar (1024 -> 1 per token) ----------------
__global__ void k_dec(const _Float16* __restrict__ h2, const float* __restrict__ dec_w,
                      const float* __restrict__ dec_b, float* __restrict__ dec){
  int wave = threadIdx.x >> 5, lane = threadIdx.x & 31;
  int tok = blockIdx.x * 8 + wave;
  const _Float16* hp = h2 + (size_t)tok * 1024;
  float s = 0.f;
  for (int i = lane; i < 1024; i += 32) s += (float)hp[i] * dec_w[i];
  for (int off = 16; off; off >>= 1) s += __shfl_xor(s, off, 32);
  if (lane == 0) dec[tok] = mishf(s + dec_b[0]);
}

// ------------------------- h = h + sigma*(h - dec) (in place) --------------
__global__ void k_h3(_Float16* __restrict__ h2, const float* __restrict__ sigma,
                     const float* __restrict__ dec, int n){
  int e = blockIdx.x * 256 + threadIdx.x;
  if (e >= n) return;
  int m = e >> 10, f = e & 1023;
  float hv = (float)h2[e];
  h2[e] = (_Float16)(hv + sigma[f] * (hv - dec[m]));
}

// ===========================================================================
extern "C" void kernel_launch(void* const* d_in, const int* in_sizes, int n_in,
                              void* d_out, int out_size, void* d_ws, size_t ws_size,
                              hipStream_t stream){
  (void)in_sizes; (void)n_in; (void)out_size; (void)ws_size;
  const float* x       = (const float*)d_in[0];
  const float* n1_g    = (const float*)d_in[1];
  const float* n1_b    = (const float*)d_in[2];
  const float* n1_m    = (const float*)d_in[3];
  const float* n1_v    = (const float*)d_in[4];
  const float* qkv_w   = (const float*)d_in[5];
  const float* cproj_w = (const float*)d_in[6];
  const float* cproj_b = (const float*)d_in[7];
  const float* aq_w    = (const float*)d_in[8];
  const float* akv_w   = (const float*)d_in[9];
  const float* aconv_w = (const float*)d_in[10];
  const float* aconv_b = (const float*)d_in[11];
  const float* an_g    = (const float*)d_in[12];
  const float* an_b    = (const float*)d_in[13];
  const float* an_m    = (const float*)d_in[14];
  const float* an_v    = (const float*)d_in[15];
  const float* aproj_w = (const float*)d_in[16];
  const float* aproj_b = (const float*)d_in[17];
  const float* n2_g    = (const float*)d_in[18];
  const float* n2_b    = (const float*)d_in[19];
  const float* n2_m    = (const float*)d_in[20];
  const float* n2_v    = (const float*)d_in[21];
  const float* fc1_w   = (const float*)d_in[22];
  const float* fc1_b   = (const float*)d_in[23];
  const float* dw_w    = (const float*)d_in[24];
  const float* dw_b    = (const float*)d_in[25];
  const float* dec_w   = (const float*)d_in[26];
  const float* dec_b   = (const float*)d_in[27];
  const float* sigma   = (const float*)d_in[28];
  const float* fc2_w   = (const float*)d_in[29];
  const float* fc2_b   = (const float*)d_in[30];
  float* out = (float*)d_out;

  const int NF = 65536;          // full-res tokens
  const int ND = 16384;          // downsampled tokens
  const int NC = 16384;          // central tokens

  char* ws = (char*)d_ws;
  size_t off = 0;
  auto alloc = [&](size_t bytes)->char*{
    char* p = ws + off; off = (off + bytes + 255) & ~(size_t)255; return p;
  };
  _Float16* xh16  = (_Float16*)alloc((size_t)NF * 256 * 2);
  _Float16* qbuf  = (_Float16*)alloc((size_t)NF * 256 * 2);
  _Float16* kvbuf = (_Float16*)alloc((size_t)ND * 512 * 2);
  _Float16* axcat = (_Float16*)alloc((size_t)ND * 1024 * 2);
  _Float16* axn   = (_Float16*)alloc((size_t)ND * 256 * 2);
  _Float16* qkvc  = (_Float16*)alloc((size_t)NC * 768 * 2);
  _Float16* gaout = (_Float16*)alloc((size_t)NF * 256 * 2);
  _Float16* caout = (_Float16*)alloc((size_t)NC * 256 * 2);
  _Float16* attnb = (_Float16*)alloc((size_t)NF * 256 * 2);
  float*    x2    = (float*)   alloc((size_t)NF * 256 * 4);
  _Float16* xn2   = (_Float16*)alloc((size_t)NF * 256 * 2);
  _Float16* h1    = (_Float16*)alloc((size_t)NF * 1024 * 2);
  _Float16* h2    = (_Float16*)alloc((size_t)NF * 1024 * 2);
  float*    decb  = (float*)   alloc((size_t)NF * 4);
  _Float16* wq    = (_Float16*)alloc(65536 * 2);
  _Float16* wkv   = (_Float16*)alloc(131072 * 2);
  _Float16* wqkv  = (_Float16*)alloc(196608 * 2);
  _Float16* wcp   = (_Float16*)alloc(65536 * 2);
  _Float16* wap   = (_Float16*)alloc(65536 * 2);
  _Float16* wfc1t = (_Float16*)alloc(262144 * 2);
  _Float16* wfc2t = (_Float16*)alloc(262144 * 2);
  _Float16* wcv   = (_Float16*)alloc(262144 * 2);
  float*    chan  = (float*)   alloc(6 * 256 * 4);

  auto blks = [](int n){ return (n + 255) / 256; };

  // channel constants + weight swizzles (into WMMA fragment layout)
  k_prep_chan<<<1, 256, 0, stream>>>(an_g, an_b, an_m, an_v, aconv_b,
                                     n1_g, n1_b, n1_m, n1_v, n2_g, n2_b, n2_m, n2_v, chan);
  k_swz<<<blks(65536),  256, 0, stream>>>(aq_w,    wq,    256, 256,  0);
  k_swz<<<blks(131072), 256, 0, stream>>>(akv_w,   wkv,   256, 512,  0);
  k_swz<<<blks(196608), 256, 0, stream>>>(qkv_w,   wqkv,  256, 768,  0);
  k_swz<<<blks(65536),  256, 0, stream>>>(cproj_w, wcp,   256, 256,  0);
  k_swz<<<blks(65536),  256, 0, stream>>>(aproj_w, wap,   256, 256,  0);
  k_swz<<<blks(262144), 256, 0, stream>>>(fc1_w,   wfc1t, 256, 1024, 1);
  k_swz<<<blks(262144), 256, 0, stream>>>(fc2_w,   wfc2t, 1024, 256, 1);
  k_swz<<<blks(262144), 256, 0, stream>>>(aconv_w, wcv,   1024, 256, 2);

  // BN1 -> NHWC f16; im2col for strided conv
  k_bn1<<<blks(NF * 256), 256, 0, stream>>>(x, chan, xh16, NF * 256);
  k_im2col<<<blks(ND * 1024), 256, 0, stream>>>(xh16, axcat, ND * 1024);

  GemmP g{};
  // conv-as-GEMM + BN(an) epilogue -> axn [ND,256]
  g = GemmP{axcat, 1024, wcv, ND, 256, 1024, 0, 1, axn, 256, nullptr,
            chan, chan + 256, nullptr, nullptr};
  k_gemm<<<dim3(ND / 64, 256 / 128), 256, 0, stream>>>(g);
  // q = xh @ aq_w -> qbuf [NF,256]
  g = GemmP{xh16, 256, wq, NF, 256, 256, 0, 0, qbuf, 256, nullptr,
            nullptr, nullptr, nullptr, nullptr};
  k_gemm<<<dim3(NF / 64, 256 / 128), 256, 0, stream>>>(g);
  // kv = axn @ akv_w -> kvbuf [ND,512]
  g = GemmP{axn, 256, wkv, ND, 512, 256, 0, 0, kvbuf, 512, nullptr,
            nullptr, nullptr, nullptr, nullptr};
  k_gemm<<<dim3(ND / 64, 512 / 128), 256, 0, stream>>>(g);
  // qkv central (gather A from central region) -> qkvc [NC,768]
  g = GemmP{xh16, 256, wqkv, NC, 768, 256, 1, 0, qkvc, 768, nullptr,
            nullptr, nullptr, nullptr, nullptr};
  k_gemm<<<dim3(NC / 64, 768 / 128), 256, 0, stream>>>(g);

  // attention cores (WMMA)
  k_mha_g<<<16 * 8 * 8, 256, 0, stream>>>(qbuf, kvbuf, gaout);
  k_mha_c<<<16 * 4 * 4 * 2, 128, 0, stream>>>(qkvc, caout);

  // output projections (central overwrites center region after global)
  g = GemmP{gaout, 256, wap, NF, 256, 256, 0, 2, attnb, 256, nullptr,
            nullptr, nullptr, aproj_b, nullptr};
  k_gemm<<<dim3(NF / 64, 256 / 128), 256, 0, stream>>>(g);
  g = GemmP{caout, 256, wcp, NC, 256, 256, 2, 2, attnb, 256, nullptr,
            nullptr, nullptr, cproj_b, nullptr};
  k_gemm<<<dim3(NC / 64, 256 / 128), 256, 0, stream>>>(g);

  // x2 = x + attn ; xn2 = BN2(x2)
  k_resid_bn2<<<blks(NF * 256), 256, 0, stream>>>(x, attnb, chan, x2, xn2, NF * 256);

  // FFN: fc1 -> depthwise+mish -> dec -> modulation -> fc2 (+resid, NCHW)
  g = GemmP{xn2, 256, wfc1t, NF, 1024, 256, 0, 2, h1, 1024, nullptr,
            nullptr, nullptr, fc1_b, nullptr};
  k_gemm<<<dim3(NF / 64, 1024 / 128), 256, 0, stream>>>(g);
  k_dw<<<blks(NF * 1024), 256, 0, stream>>>(h1, dw_w, dw_b, h2, NF * 1024);
  k_dec<<<NF / 8, 256, 0, stream>>>(h2, dec_w, dec_b, decb);
  k_h3<<<blks(NF * 1024), 256, 0, stream>>>(h2, sigma, decb, NF * 1024);
  g = GemmP{h2, 1024, wfc2t, NF, 256, 1024, 0, 3, nullptr, 256, out,
            nullptr, nullptr, fc2_b, x2};
  k_gemm<<<dim3(NF / 64, 256 / 128), 256, 0, stream>>>(g);
}
